// GCN_69853348102347
// MI455X (gfx1250) — matrix operand order repaired and checked by reading, not compiled, not run on hardware
//
#include <hip/hip_runtime.h>
#include <hip/hip_bf16.h>

// ---------------------------------------------------------------------------
// GCN (2-layer) for MI455X / gfx1250.
//  - Dense GEMMs via v_wmma_f32_16x16x32_f16 (f16 in, f32 accum), with all
//    operands pre-converted to f16 so the inner loop is load + wmma only (A b128, B 16-bit strided).
//  - Edge aggregation via wave-per-edge global_atomic_add_f32 (L2-resident:
//    all intermediates fit in the 192MB L2).
//  - Self-loop contribution folded into the bias/ReLU pass (no extra atomics).
// ---------------------------------------------------------------------------

typedef __attribute__((ext_vector_type(16))) _Float16 v16h;
typedef __attribute__((ext_vector_type(8)))  _Float16 v8h;
typedef __attribute__((ext_vector_type(4)))  _Float16 v4h;
typedef __attribute__((ext_vector_type(8)))  float    v8f;

// ---------------- init: deg=1 (self loop), zero a1 and out -----------------
__global__ void gcn_init(float* __restrict__ deg, float* __restrict__ a1,
                         float* __restrict__ outp, int N, int F1, int F2) {
    int idx = blockIdx.x * blockDim.x + threadIdx.x;
    if (idx < N * F1) a1[idx] = 0.0f;
    if (idx < N * F2) outp[idx] = 0.0f;
    if (idx < N)      deg[idx] = 1.0f;   // self-loop contributes 1 to degree
}

// ---------------- degree accumulation over edge dst ------------------------
__global__ void gcn_degree(const int* __restrict__ dst, float* __restrict__ deg, int E) {
    int e = blockIdx.x * blockDim.x + threadIdx.x;
    if (e < E) unsafeAtomicAdd(&deg[dst[e]], 1.0f);
}

// ---------------- dinv = deg^{-1/2} ----------------------------------------
__global__ void gcn_dinv(const float* __restrict__ deg, float* __restrict__ dinv, int N) {
    int i = blockIdx.x * blockDim.x + threadIdx.x;
    if (i < N) dinv[i] = rsqrtf(deg[i]);   // deg >= 1.0 always (self loop)
}

// ---------------- bulk f32 -> f16 conversion (4 elems/thread) --------------
__global__ void gcn_cvt_f16(const float* __restrict__ in, _Float16* __restrict__ out,
                            int n) {
    int i = (blockIdx.x * blockDim.x + threadIdx.x) * 4;
    if (i + 3 < n) {
        float4 t = *(const float4*)(in + i);
        v4h o;
        o[0] = (_Float16)t.x; o[1] = (_Float16)t.y;
        o[2] = (_Float16)t.z; o[3] = (_Float16)t.w;
        *(v4h*)(out + i) = o;
    } else {
        for (; i < n; ++i) out[i] = (_Float16)in[i];
    }
}

// ---------------- dense GEMM C[M,NF] = A[M,K] * B[K,NF] via WMMA -----------
// A, B already f16 row-major. One wave computes a 16-row x NF panel.
// M must be a multiple of 16 (50000 = 3125*16 holds here).
// Fragment layouts per CDNA5 ISA 7.12.2:
//   A (16x32 f16): lane&15 = row M; elems 0..7  = K k0+8*half .. +7,
//                                   elems 8..15 = K k0+16+8*half .. +7
//   B (32x16 f16): lane&15 = col N; elems K as A (k0+8*half.., k0+16+8*half..)
//   C/D (16x16 f32): VGPR r -> M = r + 8*(lane>>4); N = lane&15
template <int K, int NF>
__global__ void gcn_gemm_wmma(const _Float16* __restrict__ A,
                              const _Float16* __restrict__ B,
                              float* __restrict__ C, int M) {
    constexpr int KT = K / 32;
    constexpr int NT = NF / 16;

    int wave = (int)((blockIdx.x * blockDim.x + threadIdx.x) >> 5);
    int lane = threadIdx.x & 31;
    if (wave * 16 >= M) return;            // uniform per wave: EXEC stays all-1s

    const int half = lane >> 4;
    const int mrow = lane & 15;
    const _Float16* arow = A + (size_t)(wave * 16 + mrow) * K + half * 8;

    v8f acc[NT];
    const v8f vzero = {0.f, 0.f, 0.f, 0.f, 0.f, 0.f, 0.f, 0.f};
#pragma unroll
    for (int i = 0; i < NT; ++i) acc[i] = vzero;

#pragma unroll
    for (int kb = 0; kb < KT; ++kb) {
        const int k0 = kb * 32;

        // ---- A fragment: two 16B loads (K-contiguous halves) ----
        v8h alo = *(const v8h*)(arow + k0);
        v8h ahi = *(const v8h*)(arow + k0 + 16);
        v16h afrag;
#pragma unroll
        for (int i = 0; i < 8; ++i) { afrag[i] = alo[i]; afrag[8 + i] = ahi[i]; }

        // ---- B fragments: lane&15 = column N; elems K as A (k0+8*half.., k0+16+8*half..) ----
        const _Float16* brow = B + (size_t)(k0 + half * 8) * NF + mrow;
#pragma unroll
        for (int nt = 0; nt < NT; ++nt) {
            v16h bfrag;
#pragma unroll
            for (int i = 0; i < 8; ++i) { bfrag[i] = brow[(size_t)i * NF + nt * 16]; bfrag[8 + i] = brow[(size_t)(16 + i) * NF + nt * 16]; }

            acc[nt] = __builtin_amdgcn_wmma_f32_16x16x32_f16(
                false, afrag, false, bfrag, (short)0, acc[nt], false, false);
        }
    }

    // ---- store: lanes 0-15 write rows r, lanes 16-31 rows r+8 (coalesced) ----
#pragma unroll
    for (int nt = 0; nt < NT; ++nt) {
        const int col = nt * 16 + mrow;
#pragma unroll
        for (int r = 0; r < 8; ++r) {
            int row = wave * 16 + r + 8 * half;
            C[(size_t)row * NF + col] = acc[nt][r];
        }
    }
}

// ---------------- edge scatter: agg[dst] += h[src] * dinv[src]*dinv[dst] ----
// One wave per edge; lanes stride the F features (F=96 -> 3 iters, 64 -> 2).
__global__ void gcn_scatter(const int* __restrict__ src, const int* __restrict__ dst,
                            const float* __restrict__ dinv,
                            const float* __restrict__ h, float* __restrict__ agg,
                            int E, int F) {
    int e    = (int)((blockIdx.x * blockDim.x + threadIdx.x) >> 5);
    int lane = threadIdx.x & 31;
    if (e >= E) return;
    int s = src[e];
    int d = dst[e];
    float w = dinv[s] * dinv[d];
    const float* hs = h + (size_t)s * F;
    float* ad = agg + (size_t)d * F;
    for (int f = lane; f < F; f += 32)
        unsafeAtomicAdd(&ad[f], hs[f] * w);
}

// --- r1(f16) = relu(agg + h_self*dinv^2 + bias): feeds GEMM-2 directly -----
__global__ void gcn_bias_relu_f16(const float* __restrict__ agg,
                                  const float* __restrict__ h,
                                  const float* __restrict__ dinv,
                                  const float* __restrict__ bias,
                                  _Float16* __restrict__ outh, int N, int F) {
    int idx = blockIdx.x * blockDim.x + threadIdx.x;
    if (idx >= N * F) return;
    int i = idx / F;
    int f = idx - i * F;
    float dv = dinv[i];
    float v = agg[idx] + h[idx] * dv * dv + bias[f];
    outh[idx] = (_Float16)(v > 0.0f ? v : 0.0f);
}

// ------- out = relu(agg + h_self*dinv^2 + bias), f32 in place --------------
__global__ void gcn_bias_relu(float* __restrict__ agg, const float* __restrict__ h,
                              const float* __restrict__ dinv,
                              const float* __restrict__ bias, int N, int F) {
    int idx = blockIdx.x * blockDim.x + threadIdx.x;
    if (idx >= N * F) return;
    int i = idx / F;
    int f = idx - i * F;
    float dv = dinv[i];
    float v = agg[idx] + h[idx] * dv * dv + bias[f];
    agg[idx] = v > 0.0f ? v : 0.0f;
}

// ---------------------------------------------------------------------------
extern "C" void kernel_launch(void* const* d_in, const int* in_sizes, int n_in,
                              void* d_out, int out_size, void* d_ws, size_t ws_size,
                              hipStream_t stream) {
    const float* x   = (const float*)d_in[0];
    const int*   ei  = (const int*)d_in[1];
    const float* W1  = (const float*)d_in[2];
    const float* b1  = (const float*)d_in[3];
    const float* W2  = (const float*)d_in[4];
    const float* b2  = (const float*)d_in[5];
    float* outp = (float*)d_out;

    const int FH = in_sizes[3];            // 96 hidden
    const int FO = in_sizes[5];            // 64 out
    const int FI = in_sizes[2] / FH;       // 128 in
    const int N  = in_sizes[0] / FI;       // 50000 nodes
    const int E  = in_sizes[1] / 2;        // 800000 edges
    const int* src = ei;
    const int* dst = ei + E;

    // Workspace (all regions 16B aligned for these sizes):
    // deg[N] | dinv[N] | h1[N*FH] f32 (reused as h2[N*FO]) | a1[N*FH] f32 |
    // xh[N*FI] f16 | r1h[N*FH] f16 | w1h[FI*FH] f16 | w2h[FH*FO] f16
    char* p = (char*)d_ws;
    float* deg  = (float*)p;            p += (size_t)N * 4;
    float* dinv = (float*)p;            p += (size_t)N * 4;
    float* h1   = (float*)p;            p += (size_t)N * FH * 4;
    float* a1   = (float*)p;            p += (size_t)N * FH * 4;
    _Float16* xh  = (_Float16*)p;       p += (size_t)N * FI * 2;
    _Float16* r1h = (_Float16*)p;       p += (size_t)N * FH * 2;
    _Float16* w1h = (_Float16*)p;       p += (size_t)FI * FH * 2;
    _Float16* w2h = (_Float16*)p;       p += (size_t)FH * FO * 2;
    (void)ws_size; (void)n_in; (void)out_size;  // needs ~62MB; harness provides

    const int T = 256;
    // 1) init deg=1, zero a1, zero out
    gcn_init<<<(N * FH + T - 1) / T, T, 0, stream>>>(deg, a1, outp, N, FH, FO);
    // 2) degree accumulation
    gcn_degree<<<(E + T - 1) / T, T, 0, stream>>>(dst, deg, E);
    // 3) dinv = rsqrt(deg)
    gcn_dinv<<<(N + T - 1) / T, T, 0, stream>>>(deg, dinv, N);
    // 4) pre-convert x, W1, W2 to f16 (one-shot, HBM-streaming, ~2us total)
    gcn_cvt_f16<<<((N * FI / 4) + T - 1) / T, T, 0, stream>>>(x, xh, N * FI);
    gcn_cvt_f16<<<((FI * FH / 4) + T - 1) / T, T, 0, stream>>>(W1, w1h, FI * FH);
    gcn_cvt_f16<<<((FH * FO / 4) + T - 1) / T, T, 0, stream>>>(W2, w2h, FH * FO);
    // 5) h1 = x @ W1   (WMMA, 8 waves/block, wave per 16-row panel)
    {
        int waves = N / 16;
        gcn_gemm_wmma<128, 96><<<(waves + 7) / 8, T, 0, stream>>>(xh, w1h, h1, N);
    }
    // 6) a1[dst] += h1[src] * norm
    gcn_scatter<<<(E + 7) / 8, T, 0, stream>>>(src, dst, dinv, h1, a1, E, FH);
    // 7) r1h(f16) = relu(a1 + selfloop + b1)  — direct GEMM-2 input
    gcn_bias_relu_f16<<<(N * FH + T - 1) / T, T, 0, stream>>>(a1, h1, dinv, b1, r1h, N, FH);
    // 8) h2 = r1 @ W2   (reuse h1 buffer: N*64 <= N*96)
    {
        int waves = N / 16;
        gcn_gemm_wmma<96, 64><<<(waves + 7) / 8, T, 0, stream>>>(r1h, w2h, h1, N);
    }
    // 9) out[dst] += h2[src] * norm   (accumulate straight into d_out)
    gcn_scatter<<<(E + 7) / 8, T, 0, stream>>>(src, dst, dinv, h1, outp, E, FO);
    // 10) out = relu(out + selfloop + b2)
    gcn_bias_relu<<<(N * FO + T - 1) / T, T, 0, stream>>>(outp, h1, dinv, b2, N, FO);
}
